// Morphology_48275432407078
// MI455X (gfx1250) — matrix operand order repaired and checked by reading, not compiled, not run on hardware
//
#include <hip/hip_runtime.h>

typedef __attribute__((ext_vector_type(16))) _Float16 v16h;
typedef __attribute__((ext_vector_type(8)))  float    v8f;
typedef __attribute__((ext_vector_type(4)))  float    v4f;

#define KW      3
#define C_IN    16
#define C_OUT   32
#define H_IMG   128
#define W_IMG   128
#define CKK     144          // C_IN * KW * KW
#define KPAD    160          // padded K: 5 x 32
#define EPS     176          // ep row stride (f16), keeps 16B alignment
#define XTS     132          // xt row stride (f32)
#define LIMITV  (-100.0f)
#define SCALE   21.640425613334453f   // BETA * log2(e), BETA = 15

__global__ __launch_bounds__(256) void morph_dilate_wmma(
    const float* __restrict__ x, const float* __restrict__ w,
    float* __restrict__ out)
{
  __shared__ float xt[C_IN][3][XTS];                                  // input window
  __shared__ _Float16 __attribute__((aligned(16))) ew[C_OUT][KPAD];   // exp(B*w), A operand
  __shared__ _Float16 __attribute__((aligned(16))) ep[W_IMG][EPS];    // exp(B*(p-m)), B operand
  __shared__ float pmax[2][W_IMG];
  __shared__ float mcol[W_IMG];

  const int t   = threadIdx.x;
  const int b   = blockIdx.x >> 7;     // batch
  const int row = blockIdx.x & 127;    // image row

  // ---- Phase A: weights -> exp-space f16 (K zero-padded to 160) ----
  for (int idx = t; idx < C_OUT * KPAD; idx += 256) {
    const int o = idx / KPAD, kk = idx % KPAD;
    float v = 0.0f;
    if (kk < CKK) v = __builtin_amdgcn_exp2f(SCALE * w[o * CKK + kk]);
    ew[o][kk] = (_Float16)v;
  }
  // ---- Phase A2: stage 3 input rows with -100 halo ----
  for (int idx = t; idx < C_IN * 3 * 130; idx += 256) {
    const int c = idx / 390, rem = idx % 390;
    const int r = rem / 130, col = rem % 130;
    const int gr = row - 1 + r, gc = col - 1;
    float v = LIMITV;
    if ((unsigned)gr < H_IMG && (unsigned)gc < W_IMG)
      v = x[((b * C_IN + c) * H_IMG + gr) * W_IMG + gc];
    xt[c][r][col] = v;
  }
  __syncthreads();

  // ---- Phase B: per-column window max, then exponentials ----
  const int j    = t & 127;     // column
  const int half = t >> 7;      // channel half: c in [8*half, 8*half+8)
  const int c0   = half * 8;
  float vals[8][9];
  float lmax = -1e30f;
  #pragma unroll
  for (int cc = 0; cc < 8; ++cc)
    #pragma unroll
    for (int ki = 0; ki < 3; ++ki)
      #pragma unroll
      for (int kj = 0; kj < 3; ++kj) {
        const float v = xt[c0 + cc][ki][j + kj];
        vals[cc][ki * 3 + kj] = v;
        lmax = fmaxf(lmax, v);
      }
  pmax[half][j] = lmax;
  __syncthreads();
  const float m = fmaxf(pmax[0][j], pmax[1][j]);
  if (half == 0) mcol[j] = m;
  #pragma unroll
  for (int cc = 0; cc < 8; ++cc)
    #pragma unroll
    for (int q = 0; q < 9; ++q)
      ep[j][(c0 + cc) * 9 + q] =
          (_Float16)__builtin_amdgcn_exp2f(SCALE * (vals[cc][q] - m));
  if (half == 1)
    for (int kk = CKK; kk < EPS; ++kk) ep[j][kk] = (_Float16)0.0f;  // K pad
  __syncthreads();

  // ---- Phase C: WMMA  D(32x16) = Ew(32x160) x Ep(160x16) per wave tile ----
  const int wave = t >> 5;          // N-tile 0..7 (16 columns each)
  const int lane = t & 31;
  const int ln   = lane & 15;
  const int lh   = lane >> 4;
  const int col  = wave * 16 + ln;

  union Frag { v16h h; v4f f4[2]; };

  // B fragments: lane = column; half-wave lh picks K base; 16 contiguous f16
  Frag bf[5];
  #pragma unroll
  for (int ks = 0; ks < 5; ++ks) {
    const _Float16* p = &ep[col][ks * 32 + lh * 16];
    bf[ks].f4[0] = ((const v4f*)p)[0];
    bf[ks].f4[1] = ((const v4f*)p)[1];
  }

  v8f acc0 = {}; v8f acc1 = {};
  #pragma unroll
  for (int ks = 0; ks < 5; ++ks) {
    // A fragment: lane m = ln (+16 for tile 1); elems 0-15 -> K base+0..15,
    // base = ks*32 + lh*16
    Frag a0, a1;
    const _Float16* p0 = &ew[ln][ks * 32 + lh * 16];
    a0.f4[0] = ((const v4f*)p0)[0];
    a0.f4[1] = ((const v4f*)(p0 + 8))[0];
    const _Float16* p1 = &ew[16 + ln][ks * 32 + lh * 16];
    a1.f4[0] = ((const v4f*)p1)[0];
    a1.f4[1] = ((const v4f*)(p1 + 8))[0];
    acc0 = __builtin_amdgcn_wmma_f32_16x16x32_f16(false, a0.h, false, bf[ks].h,
                                                  (short)0, acc0, false, false);
    acc1 = __builtin_amdgcn_wmma_f32_16x16x32_f16(false, a1.h, false, bf[ks].h,
                                                  (short)0, acc1, false, false);
  }

  // ---- Epilogue: log-space result + max shift; D layout: lane half -> M rows ----
  const float mj  = mcol[col];
  const float inv = 1.0f / SCALE;
  #pragma unroll
  for (int rr = 0; rr < 8; ++rr) {
    const int o0 = lh * 8 + rr;        // M-tile 0: channels 0..15
    const int o1 = 16 + lh * 8 + rr;   // M-tile 1: channels 16..31
    const float v0 = __builtin_amdgcn_logf(acc0[rr]) * inv + mj;
    const float v1 = __builtin_amdgcn_logf(acc1[rr]) * inv + mj;
    out[((b * C_OUT + o0) * H_IMG + row) * W_IMG + col] = v0;
    out[((b * C_OUT + o1) * H_IMG + row) * W_IMG + col] = v1;
  }
}

extern "C" void kernel_launch(void* const* d_in, const int* in_sizes, int n_in,
                              void* d_out, int out_size, void* d_ws, size_t ws_size,
                              hipStream_t stream) {
  (void)n_in; (void)out_size; (void)d_ws; (void)ws_size;
  const float* x = (const float*)d_in[0];
  const float* w = (const float*)d_in[1];
  float* out = (float*)d_out;
  const int nb = in_sizes[0] / (C_IN * H_IMG * W_IMG);   // batch (=2)
  morph_dilate_wmma<<<dim3(nb * H_IMG), dim3(256), 0, stream>>>(x, w, out);
}
